// Mamba3Block_40046275068368
// MI455X (gfx1250) — matrix operand order repaired and checked by reading, not compiled, not run on hardware
//
#include <hip/hip_runtime.h>

typedef __bf16 bf16;
typedef __attribute__((ext_vector_type(16))) __bf16 v16bf;
typedef __attribute__((ext_vector_type(8)))  __bf16 v8bf;
typedef __attribute__((ext_vector_type(8)))  float  v8f;
typedef unsigned int u32x4 __attribute__((ext_vector_type(4)));
typedef int          i32x4 __attribute__((ext_vector_type(4)));
typedef int          i32x8 __attribute__((ext_vector_type(8)));

#define SEQ     2048
#define NGRP    4
#define NHEAD   16
#define NCHUNK  16
#define CHUNKSZ 128
#define JSPAD   160   // extended s axis: s = -1..127 (129 valid), padded to 160 (5 x K32)

#if defined(__AMDGCN__) && __has_builtin(__builtin_amdgcn_tensor_load_to_lds) && \
    __has_builtin(__builtin_amdgcn_s_wait_tensorcnt)
#define USE_TDM 1
#else
#define USE_TDM 0
#endif

static __device__ __forceinline__ v8f zero8() {
  v8f z;
#pragma unroll
  for (int i = 0; i < 8; i++) z[i] = 0.f;
  return z;
}

// A-fragment (16x32 bf16): per-lane K values are contiguous at p[0..7] and p[16..23]
static __device__ __forceinline__ v16bf ld_frag_a(const bf16* p) {
  v8bf lo = *(const v8bf*)p;
  v8bf hi = *(const v8bf*)(p + 16);
  return __builtin_shufflevector(lo, hi, 0, 1, 2, 3, 4, 5, 6, 7,
                                 8, 9, 10, 11, 12, 13, 14, 15);
}
// B-fragment (32x16 bf16): per-lane K values are contiguous at p[0..7] and p[16..23]
static __device__ __forceinline__ v16bf ld_frag_b(const bf16* p) {
  v8bf lo = *(const v8bf*)p;
  v8bf hi = *(const v8bf*)(p + 16);
  return __builtin_shufflevector(lo, hi, 0, 1, 2, 3, 4, 5, 6, 7,
                                 8, 9, 10, 11, 12, 13, 14, 15);
}

#if USE_TDM
// TDM descriptor group1: 2-byte elements, 2D tile (tile_dim0=32 x tile_dim1=rows),
// row stride = stride0 elements; tensor dims set huge (tiles are always in-bounds).
static __device__ __forceinline__ i32x8 tdm_g1(unsigned stride0, unsigned rows) {
  const unsigned tdim0 = 1u << 30, tdim1 = 1u << 30;
  i32x8 g;
  g[0] = (int)(1u << 16);                                       // data_size=2B, mask=0
  g[1] = (int)((tdim0 & 0xFFFFu) << 16);                        // tensor_dim0[15:0]
  g[2] = (int)(((tdim0 >> 16) & 0xFFFFu) | ((tdim1 & 0xFFFFu) << 16));
  g[3] = (int)(((tdim1 >> 16) & 0xFFFFu) | (32u << 16));        // tile_dim0 = 32
  g[4] = (int)(rows & 0xFFFFu);                                 // tile_dim1, tile_dim2=0
  g[5] = (int)stride0;                                          // dim0 stride [31:0]
  g[6] = 0;
  g[7] = 0;
  return g;
}
static __device__ __forceinline__ u32x4 tdm_g0(unsigned lds_addr, unsigned long long ga) {
  u32x4 g;
  g[0] = 1u;                                  // count=1, no gather/restore
  g[1] = lds_addr;                            // LDS byte address
  g[2] = (unsigned)ga;                        // global addr [31:0]
  g[3] = (unsigned)(ga >> 32) | (2u << 30);   // global addr [56:32] | type=2
  return g;
}
#endif

// ---------------------------------------------------------------------------
// Generic tiled bf16 WMMA GEMM: C[f32] = A[bf16 MxK] @ Bt[bf16 NxK]^T.
// Block tile 128 x NT*16; 8 waves; A tile staged at LDS 0, B tile at 8192.
// ---------------------------------------------------------------------------
template<int NT>
__global__ __launch_bounds__(256) void gemm_bf16_kernel(
    const bf16* __restrict__ A, int lda,
    const bf16* __restrict__ Bt, int ldb,
    float* __restrict__ C, int ldc, int K) {
  extern __shared__ char gsm[];
  bf16* sA  = (bf16*)gsm;           // [128][32]
  bf16* sBt = (bf16*)(gsm + 8192);  // [NT*16][32]
  const int tid = threadIdx.x, wid = tid >> 5, lane = tid & 31;
  const int mBase = blockIdx.x * 128, nBase = blockIdx.y * (NT * 16);

  v8f acc[NT];
#pragma unroll
  for (int i = 0; i < NT; i++) acc[i] = zero8();

#if USE_TDM
  const unsigned long long gA0 = (unsigned long long)(A + (size_t)mBase * lda);
  const unsigned long long gB0 = (unsigned long long)(Bt + (size_t)nBase * ldb);
  const i32x8 g1A = tdm_g1((unsigned)lda, 128u);
  const i32x8 g1B = tdm_g1((unsigned)ldb, (unsigned)(NT * 16));
  const i32x4 z4 = {0, 0, 0, 0};
  const i32x8 z8 = {0, 0, 0, 0, 0, 0, 0, 0};
#endif

  for (int k0 = 0; k0 < K; k0 += 32) {
    __syncthreads();
#if USE_TDM
    if (wid == 0) {   // one wave launches both tile DMAs (EXEC ignored by TDM)
      __builtin_amdgcn_tensor_load_to_lds(tdm_g0(0u, gA0 + (size_t)k0 * 2),
                                          g1A, z4, z4, z8, 0);
      __builtin_amdgcn_tensor_load_to_lds(tdm_g0(8192u, gB0 + (size_t)k0 * 2),
                                          g1B, z4, z4, z8, 0);
    }
    __builtin_amdgcn_s_wait_tensorcnt(0);
#else
    for (int i = tid; i < (128 * 32) / 8; i += 256) {     // 16B vector staging
      int r = i >> 2, s = i & 3;
      *(uint4*)&sA[r * 32 + s * 8] = *(const uint4*)(A + (size_t)(mBase + r) * lda + k0 + s * 8);
    }
    for (int i = tid; i < (NT * 16 * 32) / 8; i += 256) {
      int r = i >> 2, s = i & 3;
      *(uint4*)&sBt[r * 32 + s * 8] = *(const uint4*)(Bt + (size_t)(nBase + r) * ldb + k0 + s * 8);
    }
    if (k0 + 32 < K)
      __builtin_prefetch(A + (size_t)(mBase + (tid >> 1)) * lda + k0 + 32, 0, 1);
#endif
    __syncthreads();

    const int kb = (lane >> 4) * 8, kb2 = (lane >> 4) * 8;
    v16bf af = ld_frag_a(&sA[(wid * 16 + (lane & 15)) * 32 + kb]);
#pragma unroll
    for (int nt = 0; nt < NT; nt++) {
      v16bf bv = ld_frag_b(&sBt[(nt * 16 + (lane & 15)) * 32 + kb2]);
      acc[nt] = __builtin_amdgcn_wmma_f32_16x16x32_bf16(
          false, af, false, bv, (short)0, acc[nt], false, false);
    }
  }

  const int m0 = wid * 16 + ((lane >> 4) ? 8 : 0), nn = lane & 15;
#pragma unroll
  for (int nt = 0; nt < NT; nt++)
#pragma unroll
    for (int i = 0; i < 8; i++)
      C[(size_t)(mBase + m0 + i) * ldc + nBase + nt * 16 + nn] = acc[nt][i];
}

// ---------------------------------------------------------------------------
// Elementwise f32 -> bf16 convert, and transpose-convert W[KxN] -> Wt[NxK].
// ---------------------------------------------------------------------------
__global__ __launch_bounds__(256) void conv_kernel(const float* __restrict__ in,
                                                   bf16* __restrict__ out) {
  int idx = blockIdx.x * 256 + threadIdx.x;
  out[idx] = (bf16)in[idx];
}
__global__ __launch_bounds__(256) void tconv_kernel(const float* __restrict__ in, int ldin,
                                                    int rows, bf16* __restrict__ out) {
  int idx = blockIdx.x * 256 + threadIdx.x;   // rows*cols total
  int c = idx / rows, r = idx % rows;
  out[idx] = (bf16)in[(size_t)r * ldin + c];
}

// ---------------------------------------------------------------------------
// dt / lam head: proj columns 2560..2567 plus bias, softplus / sigmoid
// ---------------------------------------------------------------------------
__global__ __launch_bounds__(256) void dtlam_kernel(
    const float* __restrict__ u, const float* __restrict__ W_in,
    const float* __restrict__ b_in, float* __restrict__ dt, float* __restrict__ lam) {
  int idx = blockIdx.x * 256 + threadIdx.x;
  int row = idx >> 3, j = idx & 7;
  int col = 2560 + j;
  float acc = b_in[col];
  const float* ur = u + (size_t)row * 512;
  for (int k = 0; k < 512; k++) acc += ur[k] * W_in[(size_t)k * 2568 + col];
  if (j < 4) {
    float v = (acc > 20.f) ? acc : log1pf(__expf(acc));
    dt[row * 4 + j] = v;
  } else {
    lam[row * 4 + (j - 4)] = 1.f / (1.f + __expf(-acc));
  }
}

// ---------------------------------------------------------------------------
// Inclusive prefix sum of dt over L (per group).
// ---------------------------------------------------------------------------
__global__ __launch_bounds__(256) void cumdt_kernel(const float* __restrict__ dt,
                                                    float* __restrict__ cumdt) {
  __shared__ float s[SEQ * NGRP];
  for (int i = threadIdx.x; i < SEQ * NGRP; i += 256) s[i] = dt[i];
  __syncthreads();
  if (threadIdx.x < NGRP) {
    int g = threadIdx.x;
    float a = 0.f;
    for (int l = 0; l < SEQ; l++) { a += s[l * 4 + g]; s[l * 4 + g] = a; }
  }
  __syncthreads();
  for (int i = threadIdx.x; i < SEQ * NGRP; i += 256) cumdt[i] = s[i];
}

// ---------------------------------------------------------------------------
// Activations: xs = silu(x) (bf16); B/C rmsnorm + bias + RoPE -> Br, Cr (f32)
// ---------------------------------------------------------------------------
__global__ __launch_bounds__(256) void act_kernel(
    const float* __restrict__ proj, const float* __restrict__ cumdt,
    const float* __restrict__ theta_log, const float* __restrict__ wB,
    const float* __restrict__ wC, const float* __restrict__ bias_B,
    const float* __restrict__ bias_C, bf16* __restrict__ xs,
    float* __restrict__ Br, float* __restrict__ Cr) {
  int l = blockIdx.x;
  const float* pr = proj + (size_t)l * 2560;
  int tid = threadIdx.x;

  for (int i = tid; i < 1024; i += 256) {
    float v = pr[1024 + i];
    xs[(size_t)l * 1024 + i] = (bf16)(v / (1.f + __expf(-v)));
  }

  __shared__ float sb[256], sc[256], ssum[8];
  sb[tid] = pr[2048 + tid];
  sc[tid] = pr[2304 + tid];
  __syncthreads();
  if (tid < 8) {
    int which = tid >> 2, g = tid & 3;
    const float* src = which ? sc : sb;
    float s = 0.f;
    for (int m = 0; m < 64; m++) { float v = src[g * 64 + m]; s += v * v; }
    ssum[tid] = rsqrtf(s * (1.f / 64.f) + 1e-5f);
  }
  __syncthreads();
  {
    int g = tid >> 6, m = tid & 63;
    sb[tid] = sb[tid] * ssum[g] * wB[m] + bias_B[g * 64 + m];
    sc[tid] = sc[tid] * ssum[4 + g] * wC[m] + bias_C[g * 64 + m];
  }
  __syncthreads();
  {
    int which = tid >> 7;
    int q = tid & 127;
    int g = q >> 5, rr = q & 1, n2 = (q >> 1) & 15;
    float ang = __expf(theta_log[g * 16 + n2]) * cumdt[l * 4 + g];
    float cs = __cosf(ang), sn = __sinf(ang);
    const float* src = which ? sc : sb;
    float re = src[g * 64 + 4 * n2 + rr];
    float im = src[g * 64 + 4 * n2 + 2 + rr];
    float* dst = which ? Cr : Br;
    size_t base = (((size_t)l * 4 + g) * 32 + 2 * n2) * 2 + rr;
    dst[base]     = re * cs - im * sn;
    dst[base + 2] = re * sn + im * cs;
  }
}

// ---------------------------------------------------------------------------
// Trapezoid/decay weight (includes dt_s); js = s+1, s=-1 = prev chunk tail.
// ---------------------------------------------------------------------------
static __device__ __forceinline__ float wd_val(int t, int js, const float* cums,
                                               const float* dte, const float* lame) {
  float w = 0.f;
  if (js >= 1 && js <= 128 && (js - 1) <= t)
    w += lame[js] * dte[js] * __expf(cums[t + 1] - cums[js]);
  if (js <= t)
    w += (1.f - lame[js + 1]) * dte[js] * __expf(cums[t + 1] - cums[js + 1]);
  return w;
}

static __device__ __forceinline__ void fill_chunk_scalars(
    int c, int g, float Ag, const float* dt, const float* lam,
    const float* cumdt, float* s_cums, float* s_dt, float* s_lam) {
  int l0 = c * CHUNKSZ;
  float base = (c > 0) ? cumdt[(l0 - 1) * 4 + g] : 0.f;
  for (int j = threadIdx.x; j < 129; j += 256)
    s_cums[j] = (j == 0) ? 0.f : Ag * (cumdt[(l0 + j - 1) * 4 + g] - base);
  for (int j = threadIdx.x; j < 146; j += 256) {
    float dv = 0.f, lv = 0.f;
    if (j >= 1 && j <= 128) { dv = dt[(l0 + j - 1) * 4 + g]; lv = lam[(l0 + j - 1) * 4 + g]; }
    else if (j == 0 && c > 0) { dv = dt[(l0 - 1) * 4 + g]; lv = lam[(l0 - 1) * 4 + g]; }
    s_dt[j] = dv; s_lam[j] = lv;
  }
}

// ---------------------------------------------------------------------------
// h_final[c,h,n,p]: weighted sum over extended s of u_trap.
// ---------------------------------------------------------------------------
__global__ __launch_bounds__(256) void hfinal_kernel(
    const float* __restrict__ Br, const float* __restrict__ xup,
    const float* __restrict__ dt, const float* __restrict__ lam,
    const float* __restrict__ cumdt, const float* __restrict__ A_log,
    float* __restrict__ hfin) {
  extern __shared__ char smem_raw[];
  float* s_cums = (float*)smem_raw;
  float* s_dt   = s_cums + 132;
  float* s_lam  = s_dt + 148;
  float* s_w    = s_lam + 148;
  float* sBf    = (float*)(smem_raw + 768 * 4);   // [2][129][32]
  float* sXf    = sBf + 2 * 129 * 32;             // [2][129][64]

  int c = blockIdx.x, h = blockIdx.y, g = h >> 2;
  float Ag = -__expf(A_log[g]);
  int l0 = c * CHUNKSZ;
  fill_chunk_scalars(c, g, Ag, dt, lam, cumdt, s_cums, s_dt, s_lam);
  __syncthreads();
  for (int j = threadIdx.x; j < JSPAD; j += 256)
    s_w[j] = wd_val(127, j, s_cums, s_dt, s_lam);
  for (int idx = threadIdx.x; idx < 2 * 129 * 32; idx += 256) {
    int rp = idx / (129 * 32), rem = idx % (129 * 32);
    int js = rem / 32, n = rem % 32;
    int l = l0 + js - 1;
    sBf[(rp * 129 + js) * 32 + n] = (l >= 0) ? Br[(((size_t)l * 4 + g) * 32 + n) * 2 + rp] : 0.f;
  }
  for (int idx = threadIdx.x; idx < 2 * 129 * 64; idx += 256) {
    int rp = idx / (129 * 64), rem = idx % (129 * 64);
    int js = rem / 64, p = rem % 64;
    int l = l0 + js - 1;
    sXf[(rp * 129 + js) * 64 + p] = (l >= 0) ? xup[((size_t)l * 16 + h) * 128 + p * 2 + rp] : 0.f;
  }
  __syncthreads();

  int n = threadIdx.x >> 3, pb = (threadIdx.x & 7) * 8;
  float acc[8];
#pragma unroll
  for (int j = 0; j < 8; j++) acc[j] = 0.f;
  for (int js = 0; js < 129; js++) {
    float wv = s_w[js];
    float b0 = sBf[(0 * 129 + js) * 32 + n];
    float b1 = sBf[(1 * 129 + js) * 32 + n];
    const float* x0 = &sXf[(0 * 129 + js) * 64 + pb];
    const float* x1 = &sXf[(1 * 129 + js) * 64 + pb];
#pragma unroll
    for (int j = 0; j < 8; j++) acc[j] += wv * (b0 * x0[j] + b1 * x1[j]);
  }
  size_t base = (((size_t)c * 16 + h) * 32 + n) * 64 + pb;
#pragma unroll
  for (int j = 0; j < 8; j++) hfin[base + j] = acc[j];
}

// ---------------------------------------------------------------------------
// Sequential inter-chunk state scan (per head).
// ---------------------------------------------------------------------------
__global__ __launch_bounds__(256) void chunkscan_kernel(
    const float* __restrict__ hfin, const float* __restrict__ cumdt,
    const float* __restrict__ A_log, float* __restrict__ hst) {
  int h = blockIdx.x, g = h >> 2;
  float Ag = -__expf(A_log[g]);
  int n = threadIdx.x >> 3, pb = (threadIdx.x & 7) * 8;
  float st[8];
#pragma unroll
  for (int j = 0; j < 8; j++) st[j] = 0.f;
  for (int c = 0; c < NCHUNK; c++) {
    size_t base = (((size_t)c * 16 + h) * 32 + n) * 64 + pb;
#pragma unroll
    for (int j = 0; j < 8; j++) hst[base + j] = st[j];
    float lo = (c > 0) ? cumdt[(c * 128 - 1) * 4 + g] : 0.f;
    float d = __expf(Ag * (cumdt[(c * 128 + 127) * 4 + g] - lo));
#pragma unroll
    for (int j = 0; j < 8; j++) st[j] = st[j] * d + hfin[base + j];
  }
}

// ---------------------------------------------------------------------------
// Main SSM kernel (attention-style chunked scan), one block per (chunk, head).
// All fragment sources laid out so WMMA fragments are contiguous LDS vectors.
// ---------------------------------------------------------------------------
__global__ __launch_bounds__(256) void ssm_attn_kernel(
    const float* __restrict__ Br, const float* __restrict__ Cr,
    const float* __restrict__ xup, const float* __restrict__ dt,
    const float* __restrict__ lam, const float* __restrict__ cumdt,
    const float* __restrict__ hst, const float* __restrict__ A_log,
    bf16* __restrict__ Y) {
  extern __shared__ char smem_raw[];
  float* s_cums = (float*)smem_raw;
  float* s_dt   = s_cums + 132;
  float* s_lam  = s_dt + 148;
  bf16* sBr   = (bf16*)(smem_raw + 2048);   // [2][JSPAD][32]  (row=js, contiguous n)
  bf16* sXt   = sBr + 2 * JSPAD * 32;       // [2][64][JSPAD]  (row=p, contiguous js)
  bf16* s_hstT = sXt + 2 * 64 * JSPAD;      // [64][32]        (row=p, contiguous n)
  bf16* sC    = s_hstT + 64 * 32;           // [128][32]
  bf16* sCs   = sC + 128 * 32;              // [128][32]
  bf16* sG    = sCs + 128 * 32;             // [128][JSPAD]

  const int c = blockIdx.x, h = blockIdx.y, g = h >> 2;
  const int tid = threadIdx.x, wid = tid >> 5, lane = tid & 31;
  const int l0 = c * CHUNKSZ;
  const float Ag = -__expf(A_log[g]);

  fill_chunk_scalars(c, g, Ag, dt, lam, cumdt, s_cums, s_dt, s_lam);

  for (int idx = tid; idx < 2 * JSPAD * 32; idx += 256) {
    int rp = idx / (JSPAD * 32), rem = idx % (JSPAD * 32);
    int js = rem / 32, n = rem % 32;
    int l = l0 + js - 1;
    float v = 0.f;
    if (js <= 128 && l >= 0) v = Br[(((size_t)l * 4 + g) * 32 + n) * 2 + rp];
    sBr[(rp * JSPAD + js) * 32 + n] = (bf16)v;
  }
  for (int idx = tid; idx < 2 * 64 * JSPAD; idx += 256) {
    int rp = idx / (64 * JSPAD), rem = idx % (64 * JSPAD);
    int p = rem / JSPAD, js = rem % JSPAD;
    int l = l0 + js - 1;
    float v = 0.f;
    if (js <= 128 && l >= 0) v = xup[((size_t)l * 16 + h) * 128 + p * 2 + rp];
    sXt[(rp * 64 + p) * JSPAD + js] = (bf16)v;
  }
  for (int idx = tid; idx < 64 * 32; idx += 256) {
    int p = idx >> 5, n = idx & 31;
    s_hstT[p * 32 + n] = (bf16)hst[(((size_t)c * 16 + h) * 32 + n) * 64 + p];
  }
  __syncthreads();

  const int kb = (lane >> 4) * 8, kb2 = (lane >> 4) * 8;
  for (int r = 0; r < 2; r++) {
    for (int idx = tid; idx < 128 * 32; idx += 256) {
      int t = idx >> 5;
      float cv = Cr[((((size_t)(l0 + t)) * 4 + g) * 32 + (idx & 31)) * 2 + r];
      sC[idx]  = (bf16)cv;
      sCs[idx] = (bf16)(cv * __expf(s_cums[t + 1]));
    }
    __syncthreads();

    v8f yacc[4];
#pragma unroll
    for (int i = 0; i < 4; i++) yacc[i] = zero8();

    for (int rp = 0; rp < 2; rp++) {
      // ---- E = C @ B^T (K=32); G = WD .* E (bf16 in LDS) ----
      v16bf caf = ld_frag_a(&sC[(wid * 16 + (lane & 15)) * 32 + kb]);
#pragma unroll
      for (int nt = 0; nt < JSPAD / 16; nt++) {
        int col = nt * 16 + (lane & 15);
        v16bf bfr = ld_frag_b(&sBr[(rp * JSPAD + col) * 32 + kb2]);
        v8f ev = zero8();
        ev = __builtin_amdgcn_wmma_f32_16x16x32_bf16(false, caf, false, bfr,
                                                     (short)0, ev, false, false);
        int m0 = (lane >> 4) * 8, nn = lane & 15;
#pragma unroll
        for (int i = 0; i < 8; i++) {
          int t = wid * 16 + m0 + i, js = nt * 16 + nn;
          sG[t * JSPAD + js] = (bf16)(ev[i] * wd_val(t, js, s_cums, s_dt, s_lam));
        }
      }
      __syncthreads();

      // ---- Y += G @ Xup[rp] (K = JSPAD) ----
#pragma unroll
      for (int kt = 0; kt < JSPAD / 32; kt++) {
        v16bf gaf = ld_frag_a(&sG[(wid * 16 + (lane & 15)) * JSPAD + kt * 32 + kb]);
#pragma unroll
        for (int nt = 0; nt < 4; nt++) {
          int col = nt * 16 + (lane & 15);
          v16bf xbf = ld_frag_b(&sXt[(rp * 64 + col) * JSPAD + kt * 32 + kb2]);
          yacc[nt] = __builtin_amdgcn_wmma_f32_16x16x32_bf16(
              false, gaf, false, xbf, (short)0, yacc[nt], false, false);
        }
      }
      __syncthreads();
    }

    // ---- Y += (C * exp(cumla)) @ h_state (K=32) ----
    {
      v16bf csf = ld_frag_a(&sCs[(wid * 16 + (lane & 15)) * 32 + kb]);
#pragma unroll
      for (int nt = 0; nt < 4; nt++) {
        int col = nt * 16 + (lane & 15);
        v16bf hbf = ld_frag_b(&s_hstT[col * 32 + kb2]);
        yacc[nt] = __builtin_amdgcn_wmma_f32_16x16x32_bf16(
            false, csf, false, hbf, (short)0, yacc[nt], false, false);
      }
    }

    int m0 = (lane >> 4) * 8, nn = lane & 15;
#pragma unroll
    for (int nt = 0; nt < 4; nt++)
#pragma unroll
      for (int i = 0; i < 8; i++) {
        int t = wid * 16 + m0 + i, p = nt * 16 + nn;
        Y[((size_t)(l0 + t) * 16 + h) * 128 + p * 2 + r] = (bf16)yacc[nt][i];
      }
    __syncthreads();
  }
}

// ---------------------------------------------------------------------------
// D-skip + z gating after ydown GEMM (bf16 out for final GEMM).
// ---------------------------------------------------------------------------
__global__ __launch_bounds__(256) void gate_kernel(
    const float* __restrict__ y2, const bf16* __restrict__ xs,
    const float* __restrict__ proj, const float* __restrict__ D_param,
    bf16* __restrict__ ydz) {
  int idx = blockIdx.x * 256 + threadIdx.x;
  int l = idx >> 10, col = idx & 1023, h = col >> 6, p = col & 63;
  float v = y2[((size_t)l * 16 + h) * 64 + p] + D_param[h] * (float)xs[idx];
  float z = proj[(size_t)l * 2560 + col];
  ydz[idx] = (bf16)(v * (z / (1.f + __expf(-z))));
}

// ---------------------------------------------------------------------------
extern "C" void kernel_launch(void* const* d_in, const int* in_sizes, int n_in,
                              void* d_out, int out_size, void* d_ws, size_t ws_size,
                              hipStream_t stream) {
  const float* u         = (const float*)d_in[0];
  const float* W_in      = (const float*)d_in[1];
  const float* b_in      = (const float*)d_in[2];
  const float* W_xup     = (const float*)d_in[3];
  const float* W_ydown   = (const float*)d_in[4];
  const float* A_log     = (const float*)d_in[5];
  const float* theta_log = (const float*)d_in[6];
  const float* D_param   = (const float*)d_in[7];
  const float* wB        = (const float*)d_in[8];
  const float* wC        = (const float*)d_in[9];
  const float* bias_B    = (const float*)d_in[10];
  const float* bias_C    = (const float*)d_in[11];
  const float* W_out     = (const float*)d_in[12];
  float* out = (float*)d_out;
  float* ws  = (float*)d_ws;

  // ---- f32 workspace ----
  float* proj  = ws;                                // 2048*2560
  float* dtw   = proj + (size_t)2048 * 2560;        // 8192
  float* lamw  = dtw + 8192;
  float* cumdt = lamw + 8192;
  float* Brw   = cumdt + 8192;                      // 2048*256
  float* Crw   = Brw + (size_t)2048 * 256;
  float* xupw  = Crw + (size_t)2048 * 256;          // 32768*128
  float* hfin  = xupw + (size_t)32768 * 128;        // 256*2048
  float* hstw  = hfin + (size_t)256 * 2048;
  float* y2w   = hstw + (size_t)256 * 2048;         // 32768*64
  // ---- bf16 workspace ----
  bf16* u_bf   = (bf16*)(y2w + (size_t)32768 * 64);
  bf16* Wt_in  = u_bf + (size_t)2048 * 512;         // 2560 x 512
  bf16* xs_bf  = Wt_in + (size_t)2560 * 512;        // 32768 x 64
  bf16* Wt_xup = xs_bf + (size_t)32768 * 64;        // 128 x 64
  bf16* Y_bf   = Wt_xup + (size_t)128 * 64;         // 32768 x 128
  bf16* Wt_yd  = Y_bf + (size_t)32768 * 128;        // 64 x 128
  bf16* ydz_bf = Wt_yd + (size_t)64 * 128;          // 2048 x 1024
  bf16* Wt_out = ydz_bf + (size_t)2048 * 1024;      // 512 x 1024

  const size_t GEMM_LDS8 = 8192 + 8 * 16 * 32 * 2;  // 16384
  const size_t GEMM_LDS4 = 8192 + 4 * 16 * 32 * 2;  // 12288
  const size_t SSM_LDS = 2048 + 2 * (2 * JSPAD * 32 + 2 * 64 * JSPAD + 64 * 32 +
                                     128 * 32 + 128 * 32 + 128 * JSPAD);  // 124928
  const size_t HF_LDS = 768 * 4 + (2 * 129 * 32 + 2 * 129 * 64) * 4;     // 102144

  // 0. operand conversion / weight transposition (bf16)
  conv_kernel<<<4096, 256, 0, stream>>>(u, u_bf);
  tconv_kernel<<<5120, 256, 0, stream>>>(W_in, 2568, 512, Wt_in);
  tconv_kernel<<<32, 256, 0, stream>>>(W_xup, 128, 64, Wt_xup);
  tconv_kernel<<<32, 256, 0, stream>>>(W_ydown, 64, 128, Wt_yd);
  tconv_kernel<<<2048, 256, 0, stream>>>(W_out, 512, 1024, Wt_out);

  // 1. proj = u @ W_in[:, :2560]
  gemm_bf16_kernel<8><<<dim3(16, 20), 256, GEMM_LDS8, stream>>>(u_bf, 512, Wt_in, 512,
                                                                proj, 2560, 512);
  // 2-3. dt/lam heads, cumulative dt
  dtlam_kernel<<<64, 256, 0, stream>>>(u, W_in, b_in, dtw, lamw);
  cumdt_kernel<<<1, 256, 0, stream>>>(dtw, cumdt);
  // 4. activations, rmsnorm + RoPE
  act_kernel<<<2048, 256, 0, stream>>>(proj, cumdt, theta_log, wB, wC, bias_B, bias_C,
                                       xs_bf, Brw, Crw);
  // 5. x_up = xs(32768x64) @ W_xup
  gemm_bf16_kernel<8><<<dim3(256, 1), 256, GEMM_LDS8, stream>>>(xs_bf, 64, Wt_xup, 64,
                                                                xupw, 128, 64);
  // 6-7. chunk terminal states + inter-chunk scan
  hfinal_kernel<<<dim3(NCHUNK, NHEAD), 256, HF_LDS, stream>>>(Brw, xupw, dtw, lamw,
                                                              cumdt, A_log, hfin);
  chunkscan_kernel<<<NHEAD, 256, 0, stream>>>(hfin, cumdt, A_log, hstw);
  // 8. chunked SSM (WMMA)
  ssm_attn_kernel<<<dim3(NCHUNK, NHEAD), 256, SSM_LDS, stream>>>(
      Brw, Crw, xupw, dtw, lamw, cumdt, hstw, A_log, Y_bf);
  // 9. y2 = Y(32768x128) @ W_ydown
  gemm_bf16_kernel<4><<<dim3(256, 1), 256, GEMM_LDS4, stream>>>(Y_bf, 128, Wt_yd, 128,
                                                                y2w, 64, 128);
  // 10. D skip + silu(z) gate
  gate_kernel<<<8192, 256, 0, stream>>>(y2w, xs_bf, proj, D_param, ydz_bf);
  // 11. out = ydz(2048x1024) @ W_out
  gemm_bf16_kernel<8><<<dim3(16, 4), 256, GEMM_LDS8, stream>>>(ydz_bf, 1024, Wt_out, 1024,
                                                               out, 512, 1024);
}
